// LinearAttentionLayer_62302795596557
// MI455X (gfx1250) — matrix operand order repaired and checked by reading, not compiled, not run on hardware
//
#include <hip/hip_runtime.h>
#include <hip/hip_bf16.h>

// ---------------------------------------------------------------------------
// Types for CDNA5 WMMA
// ---------------------------------------------------------------------------
typedef __bf16 bf16_t;
typedef bf16_t v16bf __attribute__((ext_vector_type(16)));
typedef float  v8f   __attribute__((ext_vector_type(8)));

union FragBF16 {
    v16bf v;
    unsigned short u[16];
};

#define D_MODEL 1024
#define N_HEADS 16
#define D_HEAD  64
#define SEQ_B   4
#define SEQ_S   2048
#define NROWS   (SEQ_B * SEQ_S)        // 8192
#define SCALE_Q 0.125f                 // 1/sqrt(64)

// LDS tile geometry for the WMMA GEMM (row stride 72 shorts = 144B:
// 16B-aligned for b128 async writes, and 36-dword row stride => 16 distinct
// banks across the 16 fragment lanes -> conflict-free ds reads)
#define LDS_STRIDE 72
#define A_TILE_SH  (128 * LDS_STRIDE)          // 9216 shorts
#define B_TILE_SH  (64 * LDS_STRIDE)           // 4608 shorts
#define BUF_SH     (A_TILE_SH + B_TILE_SH)     // 13824 shorts per buffer

// ---------------------------------------------------------------------------
// CDNA5 async global -> LDS copy (ASYNCcnt-tracked), inline asm path
// ---------------------------------------------------------------------------
__device__ __forceinline__ void async_copy_b128(const unsigned short* gptr,
                                                unsigned short* lptr) {
    // low 32 bits of a generic LDS pointer are the LDS byte offset
    unsigned loff = (unsigned)(unsigned long long)(uintptr_t)lptr;
    asm volatile("global_load_async_to_lds_b128 %0, %1, off"
                 :: "v"(loff), "v"(gptr)
                 : "memory");
}

__device__ __forceinline__ void wait_async_zero() {
    asm volatile("s_wait_asynccnt 0" ::: "memory");
}

// ---------------------------------------------------------------------------
// fp32 -> bf16 (round-to-nearest-even via bit trick), stored as ushort
// ---------------------------------------------------------------------------
__device__ __forceinline__ unsigned short f32_to_bf16(float f) {
    unsigned int u = __float_as_uint(f);
    u += 0x7FFFu + ((u >> 16) & 1u);
    return (unsigned short)(u >> 16);
}

__global__ __launch_bounds__(256) void cvt_f32_bf16_kernel(
    const float* __restrict__ src, unsigned short* __restrict__ dst, int n) {
    int i = blockIdx.x * blockDim.x + threadIdx.x;
    int stride = gridDim.x * blockDim.x;
    for (; i < n; i += stride) dst[i] = f32_to_bf16(src[i]);
}

// ---------------------------------------------------------------------------
// NT bf16 WMMA GEMM:  C[M,N] = A[M,K] * B[N,K]^T   (both K-contiguous)
// Block tile 128(M) x 64(N), K-block 64, double-buffered async LDS staging.
// 8 waves; wave (wm 0..1, wn 0..3) computes a 64x16 strip: per K-block it runs
// 2 k-sub-steps x 4 stacked v_wmma_f32_16x16x32_bf16.
// mode 0: N=3072 fused qkv epilogue (q: phi(scale*x), k: phi(x), v: identity)
// mode 1: plain fp32 store to out0[M,N].
// ---------------------------------------------------------------------------
__global__ __launch_bounds__(256) void gemm_nt_bf16_wmma_kernel(
    const unsigned short* __restrict__ A, const unsigned short* __restrict__ B,
    float* __restrict__ out0, float* __restrict__ out1, float* __restrict__ out2,
    int M, int N, int K, int mode, float scale) {
    __shared__ unsigned short smem[2 * BUF_SH];   // 55296 B

    const int tid    = threadIdx.x;
    const int lane   = tid & 31;
    const int wave   = tid >> 5;
    const int wm     = wave >> 2;      // 0..1 -> M offset wm*64
    const int wn     = wave & 3;       // 0..3 -> N offset wn*16
    const int lane16 = lane & 15;
    const int hi     = lane >> 4;      // lane half selects K sub-range

    const int mblk = blockIdx.x * 128;
    const int nblk = blockIdx.y * 64;

    // issue one K-block of async copies into parity buffer p:
    // A 128x64 (4 chunks/thread), B 64x64 (2 chunks/thread); 16B per chunk
    auto stage = [&](int p, int kb) {
        unsigned short* dA = smem + p * BUF_SH;
        unsigned short* dB = smem + p * BUF_SH + A_TILE_SH;
#pragma unroll
        for (int i = 0; i < 4; i++) {
            int c = tid + 256 * i;           // 0..1023
            int row = c >> 3, seg = c & 7;
            async_copy_b128(A + (size_t)(mblk + row) * K + kb + seg * 8,
                            dA + row * LDS_STRIDE + seg * 8);
        }
#pragma unroll
        for (int i = 0; i < 2; i++) {
            int c = tid + 256 * i;           // 0..511
            int row = c >> 3, seg = c & 7;
            async_copy_b128(B + (size_t)(nblk + row) * K + kb + seg * 8,
                            dB + row * LDS_STRIDE + seg * 8);
        }
    };

    v8f acc[4];
#pragma unroll
    for (int i = 0; i < 4; i++) acc[i] = (v8f){};

    stage(0, 0);                             // prologue
    int pb = 0;
    for (int kb = 0; kb < K; kb += 64, pb ^= 1) {
        wait_async_zero();                   // own copies for buf[pb] done
        __syncthreads();                     // all waves' copies done; all
                                             // waves' previous reads done
        if (kb + 64 < K)
            stage(pb ^ 1, kb + 64);          // overlap next copy with compute

        const unsigned short* As = smem + pb * BUF_SH;
        const unsigned short* Bs = smem + pb * BUF_SH + A_TILE_SH;
#pragma unroll
        for (int ks = 0; ks < 64; ks += 32) {
            // B fragment: lane = N column, K = ks + hi*8 + (e&7) + (e>>3)*16
            FragBF16 bfrag;
            {
                int n  = wn * 16 + lane16;
                int k0 = ks + hi * 8;
#pragma unroll
                for (int e = 0; e < 16; e++)
                    bfrag.u[e] = Bs[n * LDS_STRIDE + k0 + (e & 7) + ((e >> 3) << 4)];
            }
#pragma unroll
            for (int mt = 0; mt < 4; mt++) {
                FragBF16 afrag;
                int m   = wm * 64 + mt * 16 + lane16;
                int kb0 = ks + hi * 8;
#pragma unroll
                for (int e = 0; e < 8; e++)
                    afrag.u[e] = As[m * LDS_STRIDE + kb0 + e];
#pragma unroll
                for (int e = 0; e < 8; e++)
                    afrag.u[8 + e] = As[m * LDS_STRIDE + 16 + kb0 + e];
                acc[mt] = __builtin_amdgcn_wmma_f32_16x16x32_bf16(
                    false, afrag.v, false, bfrag.v, (short)0, acc[mt],
                    false, false);
            }
        }
    }

    // ---- epilogue: D layout -> VGPR v is row (hi*8+v), lane16 is column
#pragma unroll
    for (int mt = 0; mt < 4; mt++) {
#pragma unroll
        for (int vv = 0; vv < 8; vv++) {
            int row = mblk + wm * 64 + mt * 16 + hi * 8 + vv;
            int col = nblk + wn * 16 + lane16;
            float val = acc[mt][vv];
            if (mode == 0) {
                if (col < 1024) {                       // q: phi(scale*x)
                    float xq = val * scale;
                    out0[(size_t)row * 1024 + col] =
                        (xq > 0.0f) ? (xq + 1.0f) : __expf(xq);
                } else if (col < 2048) {                // k: phi(x)
                    out1[(size_t)row * 1024 + (col - 1024)] =
                        (val > 0.0f) ? (val + 1.0f) : __expf(val);
                } else {                                // v: identity
                    out2[(size_t)row * 1024 + (col - 2048)] = val;
                }
            } else {
                out0[(size_t)row * N + col] = val;
            }
        }
    }
}

// ---------------------------------------------------------------------------
// gate[n,h] = sigmoid(x[n,:] . Wg[h,:] + bg[h]);  block = 16 rows x 16 heads
// ---------------------------------------------------------------------------
__global__ __launch_bounds__(256) void gate_kernel(
    const float* __restrict__ x, const float* __restrict__ Wg,
    const float* __restrict__ bg, float* __restrict__ gate) {
    int tid = threadIdx.x;
    int h   = tid & 15;
    int rl  = tid >> 4;
    int n   = blockIdx.x * 16 + rl;
    const float* xr = x + (size_t)n * D_MODEL;
    const float* wr = Wg + (size_t)h * D_MODEL;
    float s = 0.0f;
    for (int kk = 0; kk < D_MODEL; kk++) s += xr[kk] * wr[kk];
    s += bg[h];
    gate[(size_t)n * N_HEADS + h] = 1.0f / (1.0f + __expf(-s));
}

// ---------------------------------------------------------------------------
// Sequential gated scan per (b,h): S_t = g_t*S + k_t v_t^T ; y_t = q_t^T S_t
// 64 blocks (b*16+h) x 256 threads. Thread (ib=t/64, j=t%64) owns S[ib*16+ii][j].
// ---------------------------------------------------------------------------
__global__ __launch_bounds__(256) void scan_kernel(
    const float* __restrict__ q, const float* __restrict__ k,
    const float* __restrict__ v, const float* __restrict__ g,
    float* __restrict__ y) {
    const int b  = blockIdx.x >> 4;
    const int h  = blockIdx.x & 15;
    const int tid = threadIdx.x;
    const int j  = tid & 63;
    const int ib = tid >> 6;  // 0..3

    __shared__ float qs[64], ks[64], vs[64];
    __shared__ float part[4][64];
    __shared__ float gs;

    float Sreg[16];
#pragma unroll
    for (int i = 0; i < 16; i++) Sreg[i] = 0.0f;

    const size_t basev = (size_t)b * SEQ_S * D_MODEL + (size_t)h * D_HEAD;
    const size_t baseg = (size_t)b * SEQ_S * N_HEADS + h;

    for (int t = 0; t < SEQ_S; t++) {
        size_t off = basev + (size_t)t * D_MODEL;
        if (tid < 64)        qs[j] = q[off + j];
        else if (tid < 128)  ks[j] = k[off + j];
        else if (tid < 192)  vs[j] = v[off + j];
        else if (tid == 192) gs = g[baseg + (size_t)t * N_HEADS];
        __syncthreads();

        float gg = gs;
        float vj = vs[j];
        float acc = 0.0f;
#pragma unroll
        for (int ii = 0; ii < 16; ii++) {
            float kv = ks[ib * 16 + ii] * vj;
            Sreg[ii] = gg * Sreg[ii] + kv;
            acc += qs[ib * 16 + ii] * Sreg[ii];
        }
        part[ib][j] = acc;
        __syncthreads();

        if (tid < 64)
            y[off + j] = part[0][j] + part[1][j] + part[2][j] + part[3][j];
        // next-iteration LDS writes happen after the barrier above for every
        // thread's reads this iteration, so two barriers per step suffice.
    }
}

// ---------------------------------------------------------------------------
// Launch
// ---------------------------------------------------------------------------
extern "C" void kernel_launch(void* const* d_in, const int* in_sizes, int n_in,
                              void* d_out, int out_size, void* d_ws, size_t ws_size,
                              hipStream_t stream) {
    (void)in_sizes; (void)n_in; (void)out_size; (void)ws_size;

    const float* x  = (const float*)d_in[0];
    const float* Wq = (const float*)d_in[1];
    const float* Wk = (const float*)d_in[2];
    const float* Wv = (const float*)d_in[3];
    const float* Wo = (const float*)d_in[4];
    const float* Wg = (const float*)d_in[5];
    const float* bg = (const float*)d_in[6];
    float* out = (float*)d_out;

    // workspace carve-up (256B aligned slabs)
    char* ws = (char*)d_ws;
    auto carve = [&](size_t bytes) {
        char* p = ws;
        ws += (bytes + 255) & ~(size_t)255;
        return p;
    };
    const size_t nX  = (size_t)NROWS * D_MODEL;             // 8192*1024
    const size_t nW  = (size_t)D_MODEL * D_MODEL;           // 1024*1024
    unsigned short* x_bf   = (unsigned short*)carve(nX * 2);
    unsigned short* Wqkv_b = (unsigned short*)carve(3 * nW * 2);
    unsigned short* Wo_bf  = (unsigned short*)carve(nW * 2);
    float* qbuf = (float*)carve(nX * 4);
    float* kbuf = (float*)carve(nX * 4);
    float* vbuf = (float*)carve(nX * 4);
    float* gbuf = (float*)carve((size_t)NROWS * N_HEADS * 4);
    float* ybuf = (float*)carve(nX * 4);
    unsigned short* y_bf = (unsigned short*)carve(nX * 2);

    // 1) convert activations + weights to bf16
    cvt_f32_bf16_kernel<<<2048, 256, 0, stream>>>(x, x_bf, (int)nX);
    cvt_f32_bf16_kernel<<<1024, 256, 0, stream>>>(Wq, Wqkv_b + 0 * nW, (int)nW);
    cvt_f32_bf16_kernel<<<1024, 256, 0, stream>>>(Wk, Wqkv_b + 1 * nW, (int)nW);
    cvt_f32_bf16_kernel<<<1024, 256, 0, stream>>>(Wv, Wqkv_b + 2 * nW, (int)nW);
    cvt_f32_bf16_kernel<<<1024, 256, 0, stream>>>(Wo, Wo_bf, (int)nW);

    // 2) gate (fp32, small)
    gate_kernel<<<NROWS / 16, 256, 0, stream>>>(x, Wg, bg, gbuf);

    // 3) fused qkv projection + feature maps (WMMA + async LDS staging)
    gemm_nt_bf16_wmma_kernel<<<dim3(NROWS / 128, 3072 / 64), 256, 0, stream>>>(
        x_bf, Wqkv_b, qbuf, kbuf, vbuf, NROWS, 3072, D_MODEL, 0, SCALE_Q);

    // 4) gated linear-attention scan
    scan_kernel<<<SEQ_B * N_HEADS, 256, 0, stream>>>(qbuf, kbuf, vbuf, gbuf, ybuf);

    // 5) output projection (WMMA)
    cvt_f32_bf16_kernel<<<2048, 256, 0, stream>>>(ybuf, y_bf, (int)nX);
    gemm_nt_bf16_wmma_kernel<<<dim3(NROWS / 128, D_MODEL / 64), 256, 0, stream>>>(
        y_bf, Wo_bf, out, nullptr, nullptr, NROWS, D_MODEL, D_MODEL, 1, 1.0f);
}
